// TransformerEncoderBlock_32332513804419
// MI455X (gfx1250) — matrix operand order repaired and checked by reading, not compiled, not run on hardware
//
#include <hip/hip_runtime.h>
#include <hip/hip_bf16.h>
#include <math.h>

// ---------------------------------------------------------------------------
// Transformer encoder block for gfx1250 (MI455X), wave32 + WMMA f16->f32.
// Round 2: pre-transposed f16 weights ([N][K]) so every B tile stages with
// b128 loads/stores, 64x128 tile with 4 WMMAs/wave/K-step, double-buffered
// LDS with register prefetch (1 barrier per K-step), global_prefetch on the
// weight stream.
// ---------------------------------------------------------------------------

#define BM 64
#define BN 128
#define BK 32
#define THREADS 256

#define CS   197       // sequence length S
#define CSP  224       // padded S (multiple of BK) for vT / probs rows
#define CD   768       // model dim D (= DK)
#define CH   12        // heads
#define CB   16        // batch
#define CM   3072      // MLP hidden
#define CBS  (CB*CS)   // 3152 rows
#define CZ   (CB*CH)   // 192 (b,h) pairs

typedef __attribute__((ext_vector_type(16))) _Float16 v16h_t;
typedef __attribute__((ext_vector_type(8)))  _Float16 v8h_t;
typedef __attribute__((ext_vector_type(8)))  float    v8f_t;

struct BReg { v8h_t x, y; };

__device__ __forceinline__ v16h_t make_frag16(const _Float16* p, int kb)
{
    v8h_t lo = *(const v8h_t*)(p + kb);
    v8h_t hi = *(const v8h_t*)(p + kb + 16);
    v16h_t a;
    #pragma unroll
    for (int i = 0; i < 8; ++i) { a[i] = lo[i]; a[8 + i] = hi[i]; }
    return a;
}

// ---------------------------------------------------------------------------
// WMMA GEMM core. 256 threads = 8 wave32s in a 2x4 grid over a 64x128 tile;
// each wave owns 32x32 (4 x 16x16 WMMA accumulators, A/B fragment reuse).
// A staged [BM][BK] f16 row-major; B staged [BN][BK] f16 n-major so every
// fragment is contiguous LDS (ISA 7.12.2 layouts). Double-buffered LDS:
// global loads for tile i+1 issued into registers before computing tile i.
// ---------------------------------------------------------------------------
template<class ALoad, class BLoad, class CStore>
__device__ __forceinline__ void gemm_core(int K, ALoad loadA, BLoad loadB, CStore storeC)
{
    __shared__ __align__(32) _Float16 As[2][BM * BK];
    __shared__ __align__(32) _Float16 Bs[2][BN * BK];

    const int tid  = threadIdx.x;
    const int wave = tid >> 5;
    const int lane = tid & 31;
    const int wm   = wave >> 2;              // 0..1 : 32-row strip
    const int wn   = wave & 3;               // 0..3 : 32-col strip
    const int lm   = lane & 15;
    const int kbA  = (lane < 16) ? 0 : 8;    // A frag: K {0..7,16..23}|{8..15,24..31}
    const int kbB  = (lane < 16) ? 0 : 8;    // B frag: K {0..7,16..23}|{8..15,24..31}

    // staging coordinates (per thread)
    const int ar = (tid * 8) >> 5, ac = (tid * 8) & 31;   // A: 8 halfs
    const int br = tid >> 1,       bc = (tid & 1) * 16;   // B: 16 halfs

    const v8f_t z8 = {0.f,0.f,0.f,0.f,0.f,0.f,0.f,0.f};
    v8f_t c00 = z8, c01 = z8, c10 = z8, c11 = z8;

    v8h_t aReg = loadA(0, ar, ac);
    BReg  bReg = loadB(0, br, bc);
    int buf = 0;

    for (int k0 = 0; k0 < K; k0 += BK) {
        *(v8h_t*)(&As[buf][ar * BK + ac])      = aReg;
        *(v8h_t*)(&Bs[buf][br * BK + bc])      = bReg.x;
        *(v8h_t*)(&Bs[buf][br * BK + bc + 8])  = bReg.y;
        __syncthreads();

        if (k0 + BK < K) {                       // prefetch next tile into regs
            aReg = loadA(k0 + BK, ar, ac);
            bReg = loadB(k0 + BK, br, bc);
        }

        const _Float16* ap0 = &As[buf][(wm * 32 + lm) * BK];
        const _Float16* ap1 = &As[buf][(wm * 32 + 16 + lm) * BK];
        v16h_t a0 = make_frag16(ap0, kbA);
        v16h_t a1 = make_frag16(ap1, kbA);
        v16h_t b0 = make_frag16(&Bs[buf][(wn * 32 + lm) * BK], kbB);
        v16h_t b1 = make_frag16(&Bs[buf][(wn * 32 + 16 + lm) * BK], kbB);

        c00 = __builtin_amdgcn_wmma_f32_16x16x32_f16(false, a0, false, b0, (short)0, c00, false, false);
        c01 = __builtin_amdgcn_wmma_f32_16x16x32_f16(false, a0, false, b1, (short)0, c01, false, false);
        c10 = __builtin_amdgcn_wmma_f32_16x16x32_f16(false, a1, false, b0, (short)0, c10, false, false);
        c11 = __builtin_amdgcn_wmma_f32_16x16x32_f16(false, a1, false, b1, (short)0, c11, false, false);

        buf ^= 1;
    }

    // D layout (ISA 7.12.2): VGPR r -> M = r + (lane<16?0:8), N = lane&15
    const int mr = wm * 32 + ((lane < 16) ? 0 : 8);
    const int nc = wn * 32 + lm;
    #pragma unroll
    for (int r = 0; r < 8; ++r) {
        storeC(mr + r,      nc,      c00[r]);
        storeC(mr + r,      nc + 16, c01[r]);
        storeC(mr + 16 + r, nc,      c10[r]);
        storeC(mr + 16 + r, nc + 16, c11[r]);
    }
}

// ---------------------------------------------------------------------------
// Weight prep: f32 [K][N] -> f16 [N][K] (batched over z). LDS-tiled 32x32.
// ---------------------------------------------------------------------------
__global__ __launch_bounds__(THREADS)
void k_transpose_cvt(const float* __restrict__ in, _Float16* __restrict__ out,
                     int K, int N)
{
    __shared__ float t[32][33];
    const int z = blockIdx.z;
    const float* ib = in + (size_t)z * K * N;
    _Float16* ob = out + (size_t)z * K * N;
    const int k0 = blockIdx.y * 32, n0 = blockIdx.x * 32;
    const int tx = threadIdx.x & 31, ty = threadIdx.x >> 5;   // 32 x 8
    #pragma unroll
    for (int i = 0; i < 32; i += 8) {
        const int k = k0 + ty + i, n = n0 + tx;
        t[ty + i][tx] = (k < K && n < N) ? ib[(size_t)k * N + n] : 0.f;
    }
    __syncthreads();
    #pragma unroll
    for (int i = 0; i < 32; i += 8) {
        const int n = n0 + ty + i, k = k0 + tx;
        if (n < N && k < K) ob[(size_t)n * K + k] = (_Float16)t[tx][ty + i];
    }
}

// ---------------------------------------------------------------------------
// LayerNorm (population variance, eps=1e-5). One 256-thread block per row.
// ---------------------------------------------------------------------------
__global__ __launch_bounds__(THREADS)
void k_ln(const float* __restrict__ in, float* __restrict__ out,
          const float* __restrict__ g, const float* __restrict__ b, int useGB)
{
    __shared__ float rs[THREADS], rs2[THREADS];
    const int row = blockIdx.x;
    const float* x = in + (size_t)row * CD;
    float s = 0.f, s2 = 0.f;
    for (int i = threadIdx.x; i < CD; i += THREADS) { float v = x[i]; s += v; s2 += v * v; }
    rs[threadIdx.x] = s; rs2[threadIdx.x] = s2;
    __syncthreads();
    for (int off = THREADS >> 1; off > 0; off >>= 1) {
        if (threadIdx.x < off) { rs[threadIdx.x] += rs[threadIdx.x + off];
                                 rs2[threadIdx.x] += rs2[threadIdx.x + off]; }
        __syncthreads();
    }
    const float mean = rs[0] * (1.0f / CD);
    const float var  = rs2[0] * (1.0f / CD) - mean * mean;
    const float inv  = rsqrtf(var + 1e-5f);
    float* o = out + (size_t)row * CD;
    for (int i = threadIdx.x; i < CD; i += THREADS) {
        float v = (x[i] - mean) * inv;
        if (useGB) v = v * g[i] + b[i];
        o[i] = v;
    }
}

// ---------------------------------------------------------------------------
// QKV projection per (b,h): out = (xn*g_h + b_h) @ W_h + bias_h -> f16.
// storeT=0: row-major [S][DK] (q,k).  storeT=1: transposed [DK][CSP] with
// zero-filled pad rows m in [S,CSP) (v, consumed as B tile by k_attn).
// ---------------------------------------------------------------------------
__global__ __launch_bounds__(THREADS)
void k_qkv(const float* __restrict__ xn, const float* __restrict__ lng,
           const float* __restrict__ lnb, const _Float16* __restrict__ WT,
           const float* __restrict__ bias, _Float16* __restrict__ out, int storeT)
{
    const int z  = blockIdx.z;            // b*H + h
    const int bb = z / CH, h = z % CH;
    const int m0 = blockIdx.x * BM;       // rows over S
    const int n0 = blockIdx.y * BN;       // cols over DK
    const float* xb  = xn  + (size_t)bb * CS * CD;
    const float* gh  = lng + (size_t)h * CD;
    const float* bh  = lnb + (size_t)h * CD;
    const _Float16* WTh = WT + (size_t)h * CD * CD;    // [DK][D] f16
    const float* bih = bias + (size_t)h * CD;
    _Float16* oz  = out + (storeT ? (size_t)z * CD * CSP : (size_t)z * CS * CD);

    auto loadA = [&](int k0, int r, int c) -> v8h_t {
        v8h_t a;
        const int m = m0 + r;
        if (m < CS) {
            const float* xr = xb + (size_t)m * CD + k0 + c;
            const float* gr = gh + k0 + c;
            const float* br = bh + k0 + c;
            #pragma unroll
            for (int i = 0; i < 8; ++i) a[i] = (_Float16)(xr[i] * gr[i] + br[i]);
        } else {
            #pragma unroll
            for (int i = 0; i < 8; ++i) a[i] = (_Float16)0.f;
        }
        return a;
    };
    auto loadB = [&](int k0, int n, int c) -> BReg {
        const _Float16* s = WTh + (size_t)(n0 + n) * CD + k0 + c;
        __builtin_prefetch(s + 2 * BK, 0, 1);
        BReg r; r.x = *(const v8h_t*)s; r.y = *(const v8h_t*)(s + 8);
        return r;
    };
    auto storeC = [&](int ml, int nl, float v) {
        const int m = m0 + ml, n = n0 + nl;
        if (storeT == 0) {
            if (m < CS) oz[(size_t)m * CD + n] = (_Float16)(v + bih[n]);
        } else {
            if (m < CS)        oz[(size_t)n * CSP + m] = (_Float16)(v + bih[n]);
            else if (m < CSP)  oz[(size_t)n * CSP + m] = (_Float16)0.f;   // zero pad
        }
    };
    gemm_core(CD, loadA, loadB, storeC);
}

// ---------------------------------------------------------------------------
// Attention scores per (b,h): sc = (q @ k^T) / sqrt(DK) -> f32 [S][S].
// k is naturally [t][dk] = [N][K] for this GEMM.
// ---------------------------------------------------------------------------
__global__ __launch_bounds__(THREADS)
void k_scores(const _Float16* __restrict__ q, const _Float16* __restrict__ kmat,
              float* __restrict__ sc)
{
    const int z  = blockIdx.z;
    const int m0 = blockIdx.x * BM;       // s
    const int n0 = blockIdx.y * BN;       // t
    const _Float16* qz = q    + (size_t)z * CS * CD;
    const _Float16* kz = kmat + (size_t)z * CS * CD;
    float* sz = sc + (size_t)z * CS * CS;

    auto loadA = [&](int k0, int r, int c) -> v8h_t {
        const int m = m0 + r;
        if (m < CS) return *(const v8h_t*)(qz + (size_t)m * CD + k0 + c);
        v8h_t a;
        #pragma unroll
        for (int i = 0; i < 8; ++i) a[i] = (_Float16)0.f;
        return a;
    };
    auto loadB = [&](int k0, int n, int c) -> BReg {
        BReg r;
        const int t = n0 + n;
        if (t < CS) {
            const _Float16* s = kz + (size_t)t * CD + k0 + c;
            r.x = *(const v8h_t*)s; r.y = *(const v8h_t*)(s + 8);
        } else {
            #pragma unroll
            for (int i = 0; i < 8; ++i) { r.x[i] = (_Float16)0.f; r.y[i] = (_Float16)0.f; }
        }
        return r;
    };
    auto storeC = [&](int ml, int nl, float v) {
        const int m = m0 + ml, n = n0 + nl;
        if (m < CS && n < CS) sz[(size_t)m * CS + n] = v * 0.03608439182435161f; // 1/sqrt(768)
    };
    gemm_core(CD, loadA, loadB, storeC);
}

// ---------------------------------------------------------------------------
// Softmax over the QUERY axis (reference softmax(scores, axis=-2)).
// Writes probs f16 with padded row length CSP; pad columns zero-filled.
// ---------------------------------------------------------------------------
__global__ __launch_bounds__(THREADS)
void k_softmax_col(const float* __restrict__ sc, _Float16* __restrict__ pr)
{
    const int z = blockIdx.x;
    const float* s = sc + (size_t)z * CS * CS;
    _Float16*   p = pr + (size_t)z * CS * CSP;
    for (int t = threadIdx.x; t < CS; t += THREADS) {
        float mx = -3.4e38f;
        for (int i = 0; i < CS; ++i) mx = fmaxf(mx, s[(size_t)i * CS + t]);
        float sum = 0.f;
        for (int i = 0; i < CS; ++i) sum += __expf(s[(size_t)i * CS + t] - mx);
        const float inv = 1.f / sum;
        for (int i = 0; i < CS; ++i)
            p[(size_t)i * CSP + t] = (_Float16)(__expf(s[(size_t)i * CS + t] - mx) * inv);
    }
    for (int t = CS + threadIdx.x; t < CSP; t += THREADS)   // zero K-pad
        for (int i = 0; i < CS; ++i) p[(size_t)i * CSP + t] = (_Float16)0.f;
}

// ---------------------------------------------------------------------------
// attn = probs @ v per (b,h). probs [S][CSP] f16, vT [DK][CSP] f16 (both
// zero-padded to CSP in K) -> attn f16 [S][DK].
// ---------------------------------------------------------------------------
__global__ __launch_bounds__(THREADS)
void k_attn(const _Float16* __restrict__ pr, const _Float16* __restrict__ vT,
            _Float16* __restrict__ attn)
{
    const int z  = blockIdx.z;
    const int m0 = blockIdx.x * BM;       // s
    const int n0 = blockIdx.y * BN;       // dk
    const _Float16* pz = pr + (size_t)z * CS * CSP;
    const _Float16* vz = vT + (size_t)z * CD * CSP;
    _Float16* az = attn + (size_t)z * CS * CD;

    auto loadA = [&](int k0, int r, int c) -> v8h_t {
        const int m = m0 + r;
        if (m < CS) return *(const v8h_t*)(pz + (size_t)m * CSP + k0 + c);
        v8h_t a;
        #pragma unroll
        for (int i = 0; i < 8; ++i) a[i] = (_Float16)0.f;
        return a;
    };
    auto loadB = [&](int k0, int n, int c) -> BReg {
        const _Float16* s = vz + (size_t)(n0 + n) * CSP + k0 + c;
        BReg r; r.x = *(const v8h_t*)s; r.y = *(const v8h_t*)(s + 8);
        return r;
    };
    auto storeC = [&](int ml, int nl, float v) {
        const int m = m0 + ml, n = n0 + nl;
        if (m < CS) az[(size_t)m * CD + n] = (_Float16)v;
    };
    gemm_core(CSP, loadA, loadB, storeC);   // K padded; pads are zeros
}

// ---------------------------------------------------------------------------
// Output projection: x1 = concat_heads(attn) @ Wo + bo + x  (f32).
// Head-concat gather in the A loader (K = H*DK = 9216); WoT is [768][9216].
// ---------------------------------------------------------------------------
__global__ __launch_bounds__(THREADS)
void k_proj(const _Float16* __restrict__ attn, const _Float16* __restrict__ WoT,
            const float* __restrict__ bo, const float* __restrict__ x,
            float* __restrict__ x1)
{
    const int m0 = blockIdx.x * BM;       // rows over B*S
    const int n0 = blockIdx.y * BN;       // cols over D

    auto loadA = [&](int k0, int r, int c) -> v8h_t {
        const int m = m0 + r;
        if (m < CBS) {
            const int bb = m / CS, s_ = m - bb * CS;
            const int cc = k0 + c;
            const int h = cc / CD, dk = cc - h * CD;   // 8-chunk stays in one head
            return *(const v8h_t*)(attn + (((size_t)bb * CH + h) * CS + s_) * CD + dk);
        }
        v8h_t a;
        #pragma unroll
        for (int i = 0; i < 8; ++i) a[i] = (_Float16)0.f;
        return a;
    };
    auto loadB = [&](int k0, int n, int c) -> BReg {
        const _Float16* s = WoT + (size_t)(n0 + n) * (CH * CD) + k0 + c;
        __builtin_prefetch(s + 2 * BK, 0, 1);
        BReg r; r.x = *(const v8h_t*)s; r.y = *(const v8h_t*)(s + 8);
        return r;
    };
    auto storeC = [&](int ml, int nl, float v) {
        const int m = m0 + ml, n = n0 + nl;
        if (m < CBS) x1[(size_t)m * CD + n] = v + bo[n] + x[(size_t)m * CD + n];
    };
    gemm_core(CH * CD, loadA, loadB, storeC);
}

// ---------------------------------------------------------------------------
// MLP1: h = gelu(xn2 @ W1 + b1) (exact erf) -> f16.  W1T is [3072][768].
// ---------------------------------------------------------------------------
__global__ __launch_bounds__(THREADS)
void k_mlp1(const float* __restrict__ xn2, const _Float16* __restrict__ W1T,
            const float* __restrict__ b1, _Float16* __restrict__ hb)
{
    const int m0 = blockIdx.x * BM;
    const int n0 = blockIdx.y * BN;

    auto loadA = [&](int k0, int r, int c) -> v8h_t {
        v8h_t a;
        const int m = m0 + r;
        if (m < CBS) {
            const float* s = xn2 + (size_t)m * CD + k0 + c;
            #pragma unroll
            for (int i = 0; i < 8; ++i) a[i] = (_Float16)s[i];
        } else {
            #pragma unroll
            for (int i = 0; i < 8; ++i) a[i] = (_Float16)0.f;
        }
        return a;
    };
    auto loadB = [&](int k0, int n, int c) -> BReg {
        const _Float16* s = W1T + (size_t)(n0 + n) * CD + k0 + c;
        __builtin_prefetch(s + 2 * BK, 0, 1);
        BReg r; r.x = *(const v8h_t*)s; r.y = *(const v8h_t*)(s + 8);
        return r;
    };
    auto storeC = [&](int ml, int nl, float v) {
        const int m = m0 + ml, n = n0 + nl;
        if (m < CBS) {
            const float t = v + b1[n];
            const float g = 0.5f * t * (1.0f + erff(t * 0.7071067811865476f));
            hb[(size_t)m * CM + n] = (_Float16)g;
        }
    };
    gemm_core(CD, loadA, loadB, storeC);
}

// ---------------------------------------------------------------------------
// MLP2: out = h @ W2 + b2 + x1 (f32 final output).  W2T is [768][3072].
// ---------------------------------------------------------------------------
__global__ __launch_bounds__(THREADS)
void k_mlp2(const _Float16* __restrict__ hb, const _Float16* __restrict__ W2T,
            const float* __restrict__ b2, const float* __restrict__ x1,
            float* __restrict__ out)
{
    const int m0 = blockIdx.x * BM;
    const int n0 = blockIdx.y * BN;

    auto loadA = [&](int k0, int r, int c) -> v8h_t {
        const int m = m0 + r;
        if (m < CBS) return *(const v8h_t*)(hb + (size_t)m * CM + k0 + c);
        v8h_t a;
        #pragma unroll
        for (int i = 0; i < 8; ++i) a[i] = (_Float16)0.f;
        return a;
    };
    auto loadB = [&](int k0, int n, int c) -> BReg {
        const _Float16* s = W2T + (size_t)(n0 + n) * CM + k0 + c;
        __builtin_prefetch(s + 2 * BK, 0, 1);
        BReg r; r.x = *(const v8h_t*)s; r.y = *(const v8h_t*)(s + 8);
        return r;
    };
    auto storeC = [&](int ml, int nl, float v) {
        const int m = m0 + ml, n = n0 + nl;
        if (m < CBS) out[(size_t)m * CD + n] = v + b2[n] + x1[(size_t)m * CD + n];
    };
    gemm_core(CM, loadA, loadB, storeC);
}

// ---------------------------------------------------------------------------
extern "C" void kernel_launch(void* const* d_in, const int* in_sizes, int n_in,
                              void* d_out, int out_size, void* d_ws, size_t ws_size,
                              hipStream_t stream)
{
    const float* x    = (const float*)d_in[0];
    const float* ln_g = (const float*)d_in[1];
    const float* ln_b = (const float*)d_in[2];
    const float* Wq   = (const float*)d_in[3];
    const float* bq   = (const float*)d_in[4];
    const float* Wk   = (const float*)d_in[5];
    const float* bk   = (const float*)d_in[6];
    const float* Wv   = (const float*)d_in[7];
    const float* bv   = (const float*)d_in[8];
    const float* Wo   = (const float*)d_in[9];
    const float* bo   = (const float*)d_in[10];
    const float* ln2g = (const float*)d_in[11];
    const float* ln2b = (const float*)d_in[12];
    const float* W1   = (const float*)d_in[13];
    const float* b1   = (const float*)d_in[14];
    const float* W2   = (const float*)d_in[15];
    const float* b2   = (const float*)d_in[16];

    char* w = (char*)d_ws;
    size_t off = 0;
    auto alloc = [&](size_t bytes) -> char* {
        char* p = w + off;
        off += (bytes + 255) & ~(size_t)255;
        return p;
    };
    float*     xn  = (float*)    alloc((size_t)CBS * CD * 4);
    _Float16*  qb  = (_Float16*) alloc((size_t)CZ * CS * CD * 2);
    _Float16*  kb  = (_Float16*) alloc((size_t)CZ * CS * CD * 2);
    _Float16*  vT  = (_Float16*) alloc((size_t)CZ * CD * CSP * 2);
    float*     sc  = (float*)    alloc((size_t)CZ * CS * CS * 4);
    _Float16*  pr  = (_Float16*) alloc((size_t)CZ * CS * CSP * 2);
    _Float16*  at  = (_Float16*) alloc((size_t)CZ * CS * CD * 2);
    float*     x1  = (float*)    alloc((size_t)CBS * CD * 4);
    float*     xn2 = (float*)    alloc((size_t)CBS * CD * 4);
    _Float16*  hb  = (_Float16*) alloc((size_t)CBS * CM * 2);
    _Float16*  WqT = (_Float16*) alloc((size_t)CH * CD * CD * 2);
    _Float16*  WkT = (_Float16*) alloc((size_t)CH * CD * CD * 2);
    _Float16*  WvT = (_Float16*) alloc((size_t)CH * CD * CD * 2);
    _Float16*  WoT = (_Float16*) alloc((size_t)CH * CD * CD * 2);  // [768][9216]
    _Float16*  W1T = (_Float16*) alloc((size_t)CD * CM * 2);       // [3072][768]
    _Float16*  W2T = (_Float16*) alloc((size_t)CM * CD * 2);       // [768][3072]

    const int gM = (CBS + BM - 1) / BM;    // 50
    const int gS = (CS + BM - 1) / BM;     // 4

    // 0. weight prep: f32 [K][N] -> f16 [N][K]
    k_transpose_cvt<<<dim3(CD/32, CD/32, CH), THREADS, 0, stream>>>(Wq, WqT, CD, CD);
    k_transpose_cvt<<<dim3(CD/32, CD/32, CH), THREADS, 0, stream>>>(Wk, WkT, CD, CD);
    k_transpose_cvt<<<dim3(CD/32, CD/32, CH), THREADS, 0, stream>>>(Wv, WvT, CD, CD);
    k_transpose_cvt<<<dim3(CD/32, (CH*CD)/32, 1), THREADS, 0, stream>>>(Wo, WoT, CH*CD, CD);
    k_transpose_cvt<<<dim3(CM/32, CD/32, 1), THREADS, 0, stream>>>(W1, W1T, CD, CM);
    k_transpose_cvt<<<dim3(CD/32, CM/32, 1), THREADS, 0, stream>>>(W2, W2T, CM, CD);

    // 1. LN1 (raw stats; per-head gamma/beta folded into QKV A-loaders)
    k_ln<<<CBS, THREADS, 0, stream>>>(x, xn, nullptr, nullptr, 0);

    // 2. per-head Q/K/V projections (WMMA); v stored transposed+padded
    dim3 gqkv(gS, CD / BN, CZ);
    k_qkv<<<gqkv, THREADS, 0, stream>>>(xn, ln_g, ln_b, WqT, bq, qb, 0);
    k_qkv<<<gqkv, THREADS, 0, stream>>>(xn, ln_g, ln_b, WkT, bk, kb, 0);
    k_qkv<<<gqkv, THREADS, 0, stream>>>(xn, ln_g, ln_b, WvT, bv, vT, 1);

    // 3. scores = q k^T / sqrt(DK) (WMMA)
    k_scores<<<dim3(gS, (CS + BN - 1) / BN, CZ), THREADS, 0, stream>>>(qb, kb, sc);

    // 4. softmax over query axis (writes padded probs)
    k_softmax_col<<<CZ, THREADS, 0, stream>>>(sc, pr);

    // 5. attn = probs @ v (WMMA, K padded to CSP)
    k_attn<<<dim3(gS, CD / BN, CZ), THREADS, 0, stream>>>(pr, vT, at);

    // 6. x1 = concat(attn) @ Wo + bo + x (WMMA)
    k_proj<<<dim3(gM, CD / BN, 1), THREADS, 0, stream>>>(at, WoT, bo, x, x1);

    // 7. LN2 with gamma/beta
    k_ln<<<CBS, THREADS, 0, stream>>>(x1, xn2, ln2g, ln2b, 1);

    // 8. MLP
    k_mlp1<<<dim3(gM, CM / BN, 1), THREADS, 0, stream>>>(xn2, W1T, b1, hb);
    k_mlp2<<<dim3(gM, CD / BN, 1), THREADS, 0, stream>>>(hb, W2T, b2, x1, (float*)d_out);
}
